// GraphforecastPred_4681514353134
// MI455X (gfx1250) — matrix operand order repaired and checked by reading, not compiled, not run on hardware
//
#include <hip/hip_runtime.h>

typedef __attribute__((ext_vector_type(16))) _Float16 v16h;
typedef __attribute__((ext_vector_type(8)))  float    v8f;

union Frag {
    v16h h;
    unsigned int u[8];
};

__device__ __forceinline__ unsigned int pack_h2(float a, float b) {
    union { _Float16 h[2]; unsigned int u; } t;
    t.h[0] = (_Float16)a; t.h[1] = (_Float16)b;
    return t.u;
}

// ---------------------------------------------------------------------------
// Prep kernels: pack weights to f16 / rearrange for GEMM B layout (N x K)
// ---------------------------------------------------------------------------
__global__ void pack_f16(const float* __restrict__ in, _Float16* __restrict__ out, int n) {
    for (int i = blockIdx.x * blockDim.x + threadIdx.x; i < n; i += gridDim.x * blockDim.x)
        out[i] = (_Float16)in[i];
}

// W_g2 (64x128, KxN) -> (128x64, NxK) f16
__global__ void prep_wg2t(const float* __restrict__ W, _Float16* __restrict__ Wt) {
    int idx = blockIdx.x * blockDim.x + threadIdx.x;       // 8192
    if (idx >= 128 * 64) return;
    int n = idx / 64, k = idx % 64;
    Wt[idx] = (_Float16)W[k * 128 + n];
}

// conv2_w (128 x 64 x 3) -> B (128 x 192) with k = kk*64 + i
__global__ void prep_conv2w(const float* __restrict__ W, _Float16* __restrict__ Wt) {
    int idx = blockIdx.x * blockDim.x + threadIdx.x;       // 24576
    if (idx >= 128 * 192) return;
    int n = idx / 192, r = idx % 192;
    int kk = r / 64, i = r % 64;
    Wt[idx] = (_Float16)W[n * 192 + i * 3 + kk];
}

// fold conv2 bias + BN into per-channel scale/shift
__global__ void prep_bn(const float* g, const float* b, const float* m, const float* v,
                        const float* c2b, float* sc, float* sh) {
    int t = threadIdx.x;                                   // 128
    float s = g[t] * rsqrtf(v[t] + 1e-5f);
    sc[t] = s;
    sh[t] = (c2b[t] - m[t]) * s + b[t];
}

// ---------------------------------------------------------------------------
// GCN branch
// ---------------------------------------------------------------------------
// layer1: h1[m, o] = relu(b[o] + sum_n xg[m,n] W[n,o]);  xg[m,n] = mean_f x[b,n,l,f]
__global__ __launch_bounds__(64) void gcn_l1(const float* __restrict__ x,
                                             const float* __restrict__ W,
                                             const float* __restrict__ bias,
                                             _Float16* __restrict__ h1) {
    int m = blockIdx.x, t = threadIdx.x;                   // 36864 blocks x 64
    int b = m / 36, l = m % 36;
    __shared__ float xg[9];
    if (t < 9) {
        const float* p = x + ((size_t)(b * 9 + t) * 36 + l) * 7;
        float a = 0.f;
        #pragma unroll
        for (int f = 0; f < 7; f++) a += p[f];
        xg[t] = a * (1.f / 7.f);
    }
    __syncthreads();
    float acc = bias[t];
    #pragma unroll
    for (int n = 0; n < 9; n++) acc += xg[n] * W[n * 64 + t];
    h1[(size_t)m * 64 + t] = (_Float16)fmaxf(acc, 0.f);
}

// fixup rows 0..8 of h1 (agg commutes with linear map: use mean of xg rows 0..8)
__global__ __launch_bounds__(64) void gcn_fix1(const float* __restrict__ x,
                                               const float* __restrict__ W,
                                               const float* __restrict__ bias,
                                               _Float16* __restrict__ h1) {
    int t = threadIdx.x;
    __shared__ float xbar[9];
    if (t < 9) {
        float a = 0.f;
        for (int r = 0; r < 9; r++) {                      // rows m=0..8 -> b=0, l=r
            const float* p = x + ((size_t)t * 36 + r) * 7;
            float s = 0.f;
            #pragma unroll
            for (int f = 0; f < 7; f++) s += p[f];
            a += s * (1.f / 7.f);
        }
        xbar[t] = a * (1.f / 9.f);
    }
    __syncthreads();
    float acc = bias[t];
    #pragma unroll
    for (int n = 0; n < 9; n++) acc += xbar[n] * W[n * 64 + t];
    acc = fmaxf(acc, 0.f);
    for (int r = 0; r < 9; r++) h1[(size_t)r * 64 + t] = (_Float16)acc;
}

// fixup rows 0..8 of h2 (h1 rows 0..8 are identical -> mean = row 0)
__global__ __launch_bounds__(128) void gcn_fix2(const _Float16* __restrict__ h1,
                                                const float* __restrict__ W,   // 64x128 KxN
                                                const float* __restrict__ bias,
                                                float* __restrict__ h2) {
    int t = threadIdx.x;
    float acc = bias[t];
    for (int k = 0; k < 64; k++) acc += (float)h1[k] * W[k * 128 + t];
    acc = fmaxf(acc, 0.f);
    for (int r = 0; r < 9; r++) h2[(size_t)r * 128 + t] = acc;
}

// ---------------------------------------------------------------------------
// conv1: (B*NS, 7, 36) -> out1t (B*NS, 38, 64) f16, boundary rows zeroed
// ---------------------------------------------------------------------------
__global__ __launch_bounds__(256) void conv1_kernel(const float* __restrict__ x,
                                                    const float* __restrict__ w,
                                                    const float* __restrict__ bias,
                                                    _Float16* __restrict__ out) {
    int bn = blockIdx.x, t = threadIdx.x;                  // 9216 blocks x 256
    __shared__ float lx[252];
    __shared__ float lw[2240];
    __shared__ float lb[64];
    const float* xp = x + (size_t)bn * 252;
    for (int i = t; i < 252; i += 256) lx[i] = xp[i];
    for (int i = t; i < 2240; i += 256) lw[i] = w[i];
    if (t < 64) lb[t] = bias[t];
    __syncthreads();
    _Float16* op = out + (size_t)bn * 2432;
    for (int item = t; item < 2432; item += 256) {
        int l38 = item / 64, o = item % 64;
        float v = 0.f;
        if (l38 >= 1 && l38 <= 36) {
            int l = l38 - 1;
            v = lb[o];
            #pragma unroll
            for (int i = 0; i < 7; i++)
                #pragma unroll
                for (int k = 0; k < 5; k++) {
                    int p = l + k - 2;
                    if (p >= 0 && p < 36) v += lx[i * 36 + p] * lw[o * 35 + i * 5 + k];
                }
        }
        op[item] = (_Float16)v;
    }
}

// ---------------------------------------------------------------------------
// Generic WMMA GEMM: C[M,N] = epi(A[M,K] @ W[N,K]^T)
//   AT: float or _Float16 activations. ROWMAP: 0 = A + row*lda, 1 = conv im2col.
//   EPI: 1 = +bias(e0), 2 = +bias relu, 3 = *e0 + e1 (BN).  OT: float/_Float16.
// Block 128 = 4 waves; wave -> 16x64 tile; grid (M/64, N/64).
// ---------------------------------------------------------------------------
template <typename AT, int ROWMAP, int EPI, typename OT>
__global__ __launch_bounds__(128) void gemm_wmma(const AT* __restrict__ A, int lda,
                                                 const _Float16* __restrict__ W, int K,
                                                 const float* __restrict__ e0,
                                                 const float* __restrict__ e1,
                                                 OT* __restrict__ C, int ldc) {
    int tid = threadIdx.x;
    int wave = tid >> 5, lane = tid & 31;
    int m16 = lane & 15, kh = lane >> 4;
    int row0 = blockIdx.x * 64 + wave * 16;
    int col0 = blockIdx.y * 64;

    int rowA = row0 + m16;
    const AT* arow;
    if constexpr (ROWMAP == 1)
        arow = A + (size_t)(rowA / 36) * 2432 + (size_t)(rowA % 36) * 64;
    else
        arow = A + (size_t)rowA * lda;

    v8f acc[4];
    #pragma unroll
    for (int nt = 0; nt < 4; nt++) {
        float init = (EPI == 1 || EPI == 2) ? e0[col0 + nt * 16 + m16] : 0.f;
        #pragma unroll
        for (int j = 0; j < 8; j++) acc[nt][j] = init;
    }

    for (int k0 = 0; k0 < K; k0 += 32) {
        Frag a;
        if constexpr (sizeof(AT) == 4) {
            const float* p1 = (const float*)arow + k0 + kh * 8;
            float4 f1 = *(const float4*)p1;
            float4 f2 = *(const float4*)(p1 + 4);
            float4 f3 = *(const float4*)(p1 + 16);
            float4 f4 = *(const float4*)(p1 + 20);
            a.u[0] = pack_h2(f1.x, f1.y); a.u[1] = pack_h2(f1.z, f1.w);
            a.u[2] = pack_h2(f2.x, f2.y); a.u[3] = pack_h2(f2.z, f2.w);
            a.u[4] = pack_h2(f3.x, f3.y); a.u[5] = pack_h2(f3.z, f3.w);
            a.u[6] = pack_h2(f4.x, f4.y); a.u[7] = pack_h2(f4.z, f4.w);
        } else {
            const _Float16* p1 = (const _Float16*)arow + k0 + kh * 8;
            uint4 x1 = *(const uint4*)p1;
            uint4 x2 = *(const uint4*)(p1 + 16);
            a.u[0] = x1.x; a.u[1] = x1.y; a.u[2] = x1.z; a.u[3] = x1.w;
            a.u[4] = x2.x; a.u[5] = x2.y; a.u[6] = x2.z; a.u[7] = x2.w;
        }
        #pragma unroll
        for (int nt = 0; nt < 4; nt++) {
            const _Float16* bp = W + (size_t)(col0 + nt * 16 + m16) * K + k0 + kh * 8;
            Frag bf;
            uint4 y1 = *(const uint4*)bp;
            uint4 y2 = *(const uint4*)(bp + 16);
            bf.u[0] = y1.x; bf.u[1] = y1.y; bf.u[2] = y1.z; bf.u[3] = y1.w;
            bf.u[4] = y2.x; bf.u[5] = y2.y; bf.u[6] = y2.z; bf.u[7] = y2.w;
            acc[nt] = __builtin_amdgcn_wmma_f32_16x16x32_f16(
                false, a.h, false, bf.h, (short)0, acc[nt], false, false);
        }
    }

    #pragma unroll
    for (int nt = 0; nt < 4; nt++)
        #pragma unroll
        for (int j = 0; j < 8; j++) {
            int r = row0 + j + 8 * kh;
            int c = col0 + nt * 16 + m16;
            float v = acc[nt][j];
            if (EPI == 2) v = fmaxf(v, 0.f);
            if (EPI == 3) v = v * e0[c] + e1[c];
            if constexpr (sizeof(OT) == 4)
                ((float*)C)[(size_t)r * ldc + c] = v;
            else
                ((_Float16*)C)[(size_t)r * ldc + c] = (_Float16)v;
        }
}

// ---------------------------------------------------------------------------
// mean over NS=9 sensor channels: out2 (B*9*36, 128) f16 -> s (B*36, 128) f32
// ---------------------------------------------------------------------------
__global__ void mean_ns(const _Float16* __restrict__ o2, float* __restrict__ s) {
    int idx = blockIdx.x * 256 + threadIdx.x;
    if (idx >= 1024 * 36 * 128) return;
    int o = idx & 127, l = (idx >> 7) % 36, b = idx / (36 * 128);
    float acc = 0.f;
    #pragma unroll
    for (int ns = 0; ns < 9; ns++)
        acc += (float)o2[((size_t)((b * 9 + ns) * 36 + l)) * 128 + o];
    s[idx] = acc * (1.f / 9.f);
}

// ---------------------------------------------------------------------------
// attention (per batch block): qkv f16 (B*36, 384) -> o f16 (B*36, 128)
// ---------------------------------------------------------------------------
__global__ __launch_bounds__(256) void attn_kernel(const _Float16* __restrict__ qkv,
                                                   _Float16* __restrict__ o) {
    int b = blockIdx.x, t = threadIdx.x;
    __shared__ _Float16 q[36 * 128], k[36 * 128], v[36 * 128];
    __shared__ float sc[2 * 36 * 36];
    for (int idx = t; idx < 36 * 128; idx += 256) {
        int l = idx / 128, d = idx % 128;
        const _Float16* row = qkv + (size_t)(b * 36 + l) * 384;
        q[idx] = row[d]; k[idx] = row[128 + d]; v[idx] = row[256 + d];
    }
    __syncthreads();
    for (int idx = t; idx < 2 * 36 * 36; idx += 256) {
        int h = idx / 1296, qi = (idx / 36) % 36, ki = idx % 36;
        float acc = 0.f;
        #pragma unroll 8
        for (int d = 0; d < 64; d++)
            acc += (float)q[qi * 128 + h * 64 + d] * (float)k[ki * 128 + h * 64 + d];
        sc[idx] = acc * 0.125f;                            // 1/sqrt(64)
    }
    __syncthreads();
    for (int r = t; r < 72; r += 256) {
        float* row = sc + r * 36;
        float mx = row[0];
        for (int j = 1; j < 36; j++) mx = fmaxf(mx, row[j]);
        float s = 0.f;
        for (int j = 0; j < 36; j++) { float e = __expf(row[j] - mx); row[j] = e; s += e; }
        float inv = 1.f / s;
        for (int j = 0; j < 36; j++) row[j] *= inv;
    }
    __syncthreads();
    for (int idx = t; idx < 2 * 36 * 64; idx += 256) {
        int h = idx / (36 * 64), qi = (idx / 64) % 36, d = idx % 64;
        const float* row = sc + (h * 36 + qi) * 36;
        float acc = 0.f;
        #pragma unroll 6
        for (int j = 0; j < 36; j++) acc += row[j] * (float)v[j * 128 + h * 64 + d];
        o[(size_t)(b * 36 + qi) * 128 + h * 64 + d] = (_Float16)acc;
    }
}

// ---------------------------------------------------------------------------
// s = LayerNorm(s + T) * g + b, per row of 128
// ---------------------------------------------------------------------------
__global__ __launch_bounds__(128) void add_ln(float* __restrict__ S, const float* __restrict__ T,
                                              const float* __restrict__ g,
                                              const float* __restrict__ bta) {
    int r = blockIdx.x, t = threadIdx.x;
    size_t base = (size_t)r * 128;
    float v = S[base + t] + T[base + t];
    __shared__ float red[128];
    red[t] = v; __syncthreads();
    for (int st = 64; st > 0; st >>= 1) { if (t < st) red[t] += red[t + st]; __syncthreads(); }
    float mean = red[0] * (1.f / 128.f);
    __syncthreads();
    float d = v - mean;
    red[t] = d * d; __syncthreads();
    for (int st = 64; st > 0; st >>= 1) { if (t < st) red[t] += red[t + st]; __syncthreads(); }
    float var = red[0] * (1.f / 128.f);
    S[base + t] = d * rsqrtf(var + 1e-5f) * g[t] + bta[t];
}

// ---------------------------------------------------------------------------
// Fused FFN: Out = relu(S @ W1^T + b1) @ W2^T + b2   (128->2048->128)
// Block 128 = 4 waves, wave -> 16 rows. Hidden chunked by 32, LDS transpose
// between the two WMMA stages (C layout -> A layout).
// ---------------------------------------------------------------------------
__global__ __launch_bounds__(128) void ffn_fused(const float* __restrict__ S,
                                                 const _Float16* __restrict__ W1,
                                                 const float* __restrict__ b1,
                                                 const _Float16* __restrict__ W2,
                                                 const float* __restrict__ b2,
                                                 float* __restrict__ Out) {
    int tid = threadIdx.x;
    int wave = tid >> 5, lane = tid & 31;
    int m16 = lane & 15, kh = lane >> 4;
    int row0 = blockIdx.x * 64 + wave * 16;

    __shared__ _Float16 hbuf[4][16 * 32];

    // A fragments for the first GEMM (K = 128 -> 4 frags), from f32 S
    Frag af[4];
    const float* arow = S + (size_t)(row0 + m16) * 128;
    #pragma unroll
    for (int kk = 0; kk < 4; kk++) {
        const float* p1 = arow + kk * 32 + kh * 8;
        float4 f1 = *(const float4*)p1;
        float4 f2 = *(const float4*)(p1 + 4);
        float4 f3 = *(const float4*)(p1 + 16);
        float4 f4 = *(const float4*)(p1 + 20);
        af[kk].u[0] = pack_h2(f1.x, f1.y); af[kk].u[1] = pack_h2(f1.z, f1.w);
        af[kk].u[2] = pack_h2(f2.x, f2.y); af[kk].u[3] = pack_h2(f2.z, f2.w);
        af[kk].u[4] = pack_h2(f3.x, f3.y); af[kk].u[5] = pack_h2(f3.z, f3.w);
        af[kk].u[6] = pack_h2(f4.x, f4.y); af[kk].u[7] = pack_h2(f4.z, f4.w);
    }

    v8f c2[8];
    #pragma unroll
    for (int nt = 0; nt < 8; nt++) {
        float init = b2[nt * 16 + m16];
        #pragma unroll
        for (int j = 0; j < 8; j++) c2[nt][j] = init;
    }

    for (int c = 0; c < 64; c++) {                         // 64 chunks of 32 hidden units
        v8f h[2];
        #pragma unroll
        for (int nt = 0; nt < 2; nt++) {
            float init = b1[c * 32 + nt * 16 + m16];
            #pragma unroll
            for (int j = 0; j < 8; j++) h[nt][j] = init;
            #pragma unroll
            for (int kk = 0; kk < 4; kk++) {
                const _Float16* bp = W1 + (size_t)(c * 32 + nt * 16 + m16) * 128 + kk * 32 + kh * 8;
                Frag bf;
                uint4 y1 = *(const uint4*)bp;
                uint4 y2 = *(const uint4*)(bp + 16);
                bf.u[0] = y1.x; bf.u[1] = y1.y; bf.u[2] = y1.z; bf.u[3] = y1.w;
                bf.u[4] = y2.x; bf.u[5] = y2.y; bf.u[6] = y2.z; bf.u[7] = y2.w;
                h[nt] = __builtin_amdgcn_wmma_f32_16x16x32_f16(
                    false, af[kk].h, false, bf.h, (short)0, h[nt], false, false);
            }
        }
        // relu + store hidden tile (16x32) to LDS in row-major f16
        #pragma unroll
        for (int nt = 0; nt < 2; nt++)
            #pragma unroll
            for (int j = 0; j < 8; j++)
                hbuf[wave][(j + 8 * kh) * 32 + nt * 16 + m16] = (_Float16)fmaxf(h[nt][j], 0.f);
        __syncthreads();
        // reload as A fragment (16x32)
        Frag a2;
        {
            const _Float16* p1 = &hbuf[wave][m16 * 32 + kh * 8];
            uint4 x1 = *(const uint4*)p1;
            uint4 x2 = *(const uint4*)(p1 + 16);
            a2.u[0] = x1.x; a2.u[1] = x1.y; a2.u[2] = x1.z; a2.u[3] = x1.w;
            a2.u[4] = x2.x; a2.u[5] = x2.y; a2.u[6] = x2.z; a2.u[7] = x2.w;
        }
        #pragma unroll
        for (int nt = 0; nt < 8; nt++) {
            const _Float16* bp = W2 + (size_t)(nt * 16 + m16) * 2048 + c * 32 + kh * 8;
            Frag bf;
            uint4 y1 = *(const uint4*)bp;
            uint4 y2 = *(const uint4*)(bp + 16);
            bf.u[0] = y1.x; bf.u[1] = y1.y; bf.u[2] = y1.z; bf.u[3] = y1.w;
            bf.u[4] = y2.x; bf.u[5] = y2.y; bf.u[6] = y2.z; bf.u[7] = y2.w;
            c2[nt] = __builtin_amdgcn_wmma_f32_16x16x32_f16(
                false, a2.h, false, bf.h, (short)0, c2[nt], false, false);
        }
        __syncthreads();
    }

    #pragma unroll
    for (int nt = 0; nt < 8; nt++)
        #pragma unroll
        for (int j = 0; j < 8; j++)
            Out[(size_t)(row0 + j + 8 * kh) * 128 + nt * 16 + m16] = c2[nt][j];
}

// ---------------------------------------------------------------------------
// fuse branches + attention pooling + head (+ LeakyReLU 0.1)
// ---------------------------------------------------------------------------
__global__ __launch_bounds__(128) void final_kernel(const float* __restrict__ s,
                                                    const float* __restrict__ xgcn,
                                                    const float* __restrict__ attn_w,
                                                    const float* __restrict__ attn_b,
                                                    const float* __restrict__ fc_w,
                                                    const float* __restrict__ fc_b,
                                                    float* __restrict__ out) {
    int b = blockIdx.x, t = threadIdx.x;
    __shared__ float sf[36 * 128];
    __shared__ float logit[36];
    __shared__ float red[128];
    size_t base = (size_t)b * 36 * 128;
    for (int idx = t; idx < 36 * 128; idx += 128)
        sf[idx] = s[base + idx] + xgcn[base + idx];
    __syncthreads();
    if (t < 36) {
        float a = attn_b[0];
        for (int d = 0; d < 128; d++) a += sf[t * 128 + d] * attn_w[d];
        logit[t] = a;
    }
    __syncthreads();
    if (t == 0) {
        float mx = logit[0];
        for (int j = 1; j < 36; j++) mx = fmaxf(mx, logit[j]);
        float ssum = 0.f;
        for (int j = 0; j < 36; j++) { float e = __expf(logit[j] - mx); logit[j] = e; ssum += e; }
        float inv = 1.f / ssum;
        for (int j = 0; j < 36; j++) logit[j] *= inv;
    }
    __syncthreads();
    float pooled = 0.f;
    for (int l = 0; l < 36; l++) pooled += logit[l] * sf[l * 128 + t];
    red[t] = pooled * fc_w[t];
    __syncthreads();
    for (int st = 64; st > 0; st >>= 1) { if (t < st) red[t] += red[t + st]; __syncthreads(); }
    if (t == 0) {
        float o = red[0] + fc_b[0];
        out[b] = (o >= 0.f) ? o : 0.1f * o;
    }
}

// ---------------------------------------------------------------------------
extern "C" void kernel_launch(void* const* d_in, const int* in_sizes, int n_in,
                              void* d_out, int out_size, void* d_ws, size_t ws_size,
                              hipStream_t stream) {
    (void)in_sizes; (void)n_in; (void)out_size; (void)ws_size;
    const float* x    = (const float*)d_in[0];
    const float* W_g1 = (const float*)d_in[1];  const float* b_g1 = (const float*)d_in[2];
    const float* W_g2 = (const float*)d_in[3];  const float* b_g2 = (const float*)d_in[4];
    const float* c1w  = (const float*)d_in[5];  const float* c1b  = (const float*)d_in[6];
    const float* c2w  = (const float*)d_in[7];  const float* c2b  = (const float*)d_in[8];
    const float* bng  = (const float*)d_in[9];  const float* bnb  = (const float*)d_in[10];
    const float* bnm  = (const float*)d_in[11]; const float* bnv  = (const float*)d_in[12];
    const float* ipw  = (const float*)d_in[13]; const float* ipb  = (const float*)d_in[14];
    const float* opw  = (const float*)d_in[15]; const float* opb  = (const float*)d_in[16];
    const float* l1g  = (const float*)d_in[17]; const float* l1b  = (const float*)d_in[18];
    const float* l2g  = (const float*)d_in[19]; const float* l2b  = (const float*)d_in[20];
    const float* f1w  = (const float*)d_in[21]; const float* f1b  = (const float*)d_in[22];
    const float* f2w  = (const float*)d_in[23]; const float* f2b  = (const float*)d_in[24];
    const float* aw   = (const float*)d_in[25]; const float* ab   = (const float*)d_in[26];
    const float* fw   = (const float*)d_in[27]; const float* fb   = (const float*)d_in[28];

    char* ws = (char*)d_ws; size_t off = 0;
    auto alloc = [&](size_t bytes) -> void* {
        off = (off + 255) & ~(size_t)255;
        void* p = ws + off; off += bytes; return p;
    };
    _Float16* wIP  = (_Float16*)alloc((size_t)2 * 384 * 128 * 2);
    _Float16* wOP  = (_Float16*)alloc((size_t)2 * 128 * 128 * 2);
    _Float16* wF1  = (_Float16*)alloc((size_t)2 * 2048 * 128 * 2);
    _Float16* wF2  = (_Float16*)alloc((size_t)2 * 128 * 2048 * 2);
    _Float16* wG2t = (_Float16*)alloc((size_t)128 * 64 * 2);
    _Float16* wC2  = (_Float16*)alloc((size_t)128 * 192 * 2);
    float*    bnsc = (float*)alloc(128 * 4);
    float*    bnsh = (float*)alloc(128 * 4);
    _Float16* h1   = (_Float16*)alloc((size_t)36864 * 64 * 2);
    float*    h2   = (float*)alloc((size_t)36864 * 128 * 4);     // x_gcn
    float*    sbuf = (float*)alloc((size_t)36864 * 128 * 4);
    char*     region1 = (char*)alloc((size_t)9216 * 38 * 64 * 2);    // out1t, later qkv+o
    char*     region2 = (char*)alloc((size_t)331776 * 128 * 2);      // out2, later tmp
    _Float16* out1t = (_Float16*)region1;
    _Float16* out2  = (_Float16*)region2;
    _Float16* qkv16 = (_Float16*)region1;                            // 28.3 MB
    _Float16* o16   = (_Float16*)(region1 + (size_t)36864 * 384 * 2);// +9.4 MB (<44.8)
    float*    tmp   = (float*)region2;                               // 18.9 MB (<85)

    // ---- prep ----
    pack_f16<<<384, 256, 0, stream>>>(ipw, wIP, 2 * 384 * 128);
    pack_f16<<<128, 256, 0, stream>>>(opw, wOP, 2 * 128 * 128);
    pack_f16<<<2048, 256, 0, stream>>>(f1w, wF1, 2 * 2048 * 128);
    pack_f16<<<2048, 256, 0, stream>>>(f2w, wF2, 2 * 128 * 2048);
    prep_wg2t<<<32, 256, 0, stream>>>(W_g2, wG2t);
    prep_conv2w<<<96, 256, 0, stream>>>(c2w, wC2);
    prep_bn<<<1, 128, 0, stream>>>(bng, bnb, bnm, bnv, c2b, bnsc, bnsh);

    // ---- GCN branch ----
    gcn_l1<<<36864, 64, 0, stream>>>(x, W_g1, b_g1, h1);
    gcn_fix1<<<1, 64, 0, stream>>>(x, W_g1, b_g1, h1);
    gemm_wmma<_Float16, 0, 2, float><<<dim3(576, 2), 128, 0, stream>>>(
        h1, 64, wG2t, 64, b_g2, nullptr, h2, 128);
    gcn_fix2<<<1, 128, 0, stream>>>(h1, W_g2, b_g2, h2);

    // ---- conv branch ----
    conv1_kernel<<<9216, 256, 0, stream>>>(x, c1w, c1b, out1t);
    gemm_wmma<_Float16, 1, 3, _Float16><<<dim3(5184, 2), 128, 0, stream>>>(
        out1t, 0, wC2, 192, bnsc, bnsh, out2, 128);
    mean_ns<<<(1024 * 36 * 128 + 255) / 256, 256, 0, stream>>>(out2, sbuf);

    // ---- transformer (2 layers) ----
    for (int i = 0; i < 2; i++) {
        gemm_wmma<float, 0, 1, _Float16><<<dim3(576, 6), 128, 0, stream>>>(
            sbuf, 128, wIP + (size_t)i * 384 * 128, 128, ipb + i * 384, nullptr, qkv16, 384);
        attn_kernel<<<1024, 256, 0, stream>>>(qkv16, o16);
        gemm_wmma<_Float16, 0, 1, float><<<dim3(576, 2), 128, 0, stream>>>(
            o16, 128, wOP + (size_t)i * 128 * 128, 128, opb + i * 128, nullptr, tmp, 128);
        add_ln<<<36864, 128, 0, stream>>>(sbuf, tmp, l1g + i * 128, l1b + i * 128);
        ffn_fused<<<576, 128, 0, stream>>>(sbuf, wF1 + (size_t)i * 2048 * 128, f1b + i * 2048,
                                           wF2 + (size_t)i * 128 * 2048, f2b + i * 128, tmp);
        add_ln<<<36864, 128, 0, stream>>>(sbuf, tmp, l2g + i * 128, l2b + i * 128);
    }

    // ---- fuse + pool + head ----
    final_kernel<<<1024, 128, 0, stream>>>(sbuf, h2, aw, ab, fw, fb, (float*)d_out);
}
